// COR_Critic_21998822490464
// MI455X (gfx1250) — compile-verified, matrix operand order checked by reading, not run on hardware
//
#include <hip/hip_runtime.h>
#include <hip/hip_bf16.h>

typedef __attribute__((ext_vector_type(16))) __bf16 v16bf;
typedef __attribute__((ext_vector_type(8)))  float  v8f;

union AFrag { v16bf v; uint4 u[2]; };

#define ROWS      64
#define XSTRIDE   1032          // bf16 elems per activation row (1024 + 8 pad, 16B-aligned stride)
#define H1STRIDE  264           // bf16 elems per H1 row (256 + 8 pad)

// ---- fast transcendentals (gfx1250 has native v_tanh_f32 / v_rcp_f32) ------
__device__ __forceinline__ float fast_tanhf(float x) {
#if __has_builtin(__builtin_amdgcn_tanhf)
    return __builtin_amdgcn_tanhf(x);
#else
    // branch-free fallback: tanh(|x|) = (1 - e) / (1 + e), e = exp(-2|x|)
    float ax = __builtin_fabsf(x);
    float e  = __expf(-2.0f * ax);
    float t  = (1.0f - e) * __builtin_amdgcn_rcpf(1.0f + e);
    return __builtin_copysignf(t, x);
#endif
}

__device__ __forceinline__ float fast_sigmoid(float x) {
    return __builtin_amdgcn_rcpf(1.0f + __expf(-x));
}

// Element -> K mapping inside a 16-bit A/B fragment (CDNA5 16x32 layout):
// elements 0..7 hold K = hi*8 + e, elements 8..15 hold K = 16 + hi*8 + (e-8)
__device__ __forceinline__ int kmap(int e, int hi) {
    return (e < 8) ? (hi * 8 + e) : (16 + hi * 8 + (e - 8));
}

// ---------------------------------------------------------------------------
// Pack all weights into bf16, fragment-major layout:
//   frag index f -> 512 bf16 at pX[f*512 + lane*16 + e]; lane loads 32 contiguous bytes.
// ---------------------------------------------------------------------------
__global__ __launch_bounds__(256) void pack_weights(
    const float* __restrict__ W1, const float* __restrict__ W2,
    const float* __restrict__ Wq1, const float* __restrict__ Wq2,
    __bf16* __restrict__ pW1, __bf16* __restrict__ pW2,
    __bf16* __restrict__ pWq1, __bf16* __restrict__ pWq2)
{
    const long gid = (long)blockIdx.x * 256 + threadIdx.x;
    const long E1 = 64L   * 512;   // 32768
    const long E2 = 2048L * 512;   // 1048576
    const long E3 = 512L  * 512;   // 262144
    const long E4 = 64L   * 512;   // 32768
    if (gid >= E1 + E2 + E3 + E4) return;

    long l; int frag, rem, lane, e, hi, r;
    if (gid < E1) {
        l = gid; frag = (int)(l >> 9); rem = (int)(l & 511);
        lane = rem >> 4; e = rem & 15; hi = lane >> 4; r = lane & 15;
        int k = kmap(e, hi);
        int n = frag * 16 + r;
        int h = n >> 5, kk = n & 31;
        pW1[l] = (__bf16)W1[h * 1024 + k * 32 + kk];
    } else if (gid < E1 + E2) {
        l = gid - E1; frag = (int)(l >> 9); rem = (int)(l & 511);
        lane = rem >> 4; e = rem & 15; hi = lane >> 4; r = lane & 15;
        int nT = frag >> 5, kT = frag & 31;
        int d = kT * 32 + kmap(e, hi);
        int n = nT * 16 + r;
        int h = n >> 5, kk = n & 31;
        pW2[l] = (__bf16)W2[h * 32768 + d * 32 + kk];
    } else if (gid < E1 + E2 + E3) {
        l = gid - E1 - E2; frag = (int)(l >> 9); rem = (int)(l & 511);
        lane = rem >> 4; e = rem & 15; hi = lane >> 4; r = lane & 15;
        int nT = frag >> 5, kT = frag & 31;
        int d = kT * 32 + kmap(e, hi);
        int n = nT * 16 + r;
        pWq1[l] = (__bf16)Wq1[d * 256 + n];
    } else {
        l = gid - E1 - E2 - E3; frag = (int)(l >> 9); rem = (int)(l & 511);
        lane = rem >> 4; e = rem & 15; hi = lane >> 4; r = lane & 15;
        int nT = frag >> 3, kT = frag & 7;
        int d = kT * 32 + kmap(e, hi);
        int n = nT * 16 + r;
        pWq2[l] = (__bf16)Wq2[d * 128 + n];
    }
}

// ---------------------------------------------------------------------------
// Fused critic: 64 rows per workgroup, 8 waves, activations resident in LDS,
// bf16 WMMA with f32 accumulation, native v_tanh epilogues, parallel LN.
// ---------------------------------------------------------------------------
__global__ __launch_bounds__(256) void critic_fused(
    const float* __restrict__ state, const float* __restrict__ action,
    const float* __restrict__ b1, const float* __restrict__ g1,
    const float* __restrict__ b2, const float* __restrict__ g2,
    const float* __restrict__ bq1, const float* __restrict__ ln1g, const float* __restrict__ ln1b,
    const float* __restrict__ bq2, const float* __restrict__ ln2g, const float* __restrict__ ln2b,
    const float* __restrict__ Wq3, const float* __restrict__ bq3,
    const __bf16* __restrict__ pW1, const __bf16* __restrict__ pW2,
    const __bf16* __restrict__ pWq1, const __bf16* __restrict__ pWq2,
    float* __restrict__ out)
{
    __shared__ __align__(16) unsigned char smem[268288];
    __bf16* X1  = (__bf16*)(smem);                  // [64][1032] bf16 (132096 B)
    __bf16* X2  = (__bf16*)(smem + 132096);         // [64][1032] bf16
    __bf16* SA  = (__bf16*)(smem + 264192);         // [64][32]   bf16 (4096 B)
    float*  H1f = (float*)(smem);                   // [64][256] f32 overlays X1
    __bf16* H1  = (__bf16*)(smem + 65536);          // [64][264] bf16
    float*  H2f = (float*)(smem + 99328);           // [64][128] f32 -> ends at 132096
    float*  red = (float*)(smem + 264192);          // [64][4][2] f32 scratch overlays SA
    float*  stats = (float*)(smem + 264192 + 2048); // [64][2] f32 (mu, rs)

    const int tid  = threadIdx.x;
    const int wave = tid >> 5;
    const int lane = tid & 31;
    const int r    = lane & 15;      // row (A) / column (B,C,D) within a tile
    const int hi   = lane >> 4;
    const long row0 = (long)blockIdx.x * ROWS;

    // ---- stage 0: stage concat(state, action) into LDS as bf16 -------------
    for (int i = tid; i < ROWS * 32; i += 256) {
        int rr = i >> 5, c = i & 31;
        float v = (c < 24) ? state[(row0 + rr) * 24 + c]
                           : action[(row0 + rr) * 8 + (c - 24)];
        SA[rr * 32 + c] = (__bf16)v;
    }
    __syncthreads();

    // ---- stage 1: X1 = gate1 * tanh( SA[64x32] @ W1'[32x1024] + b1 ) -------
    for (int t = wave; t < 256; t += 8) {           // 256 tiles: 4 M x 64 N
        int m = t >> 6, n = t & 63;
        AFrag a, b;
        const unsigned char* abase = (const unsigned char*)SA + (m * 16 + r) * 64 + hi * 16;
        a.u[0] = *(const uint4*)(abase);
        a.u[1] = *(const uint4*)(abase + 32);
        const uint4* bb = (const uint4*)(pW1 + ((long)n * 512 + lane * 16));
        b.u[0] = bb[0]; b.u[1] = bb[1];
        v8f c = {0,0,0,0,0,0,0,0};
        c = __builtin_amdgcn_wmma_f32_16x16x32_bf16(false, a.v, false, b.v, (short)0, c, false, false);
        int col = n * 16 + r;
        float bias = b1[col];
        float gate = fast_sigmoid(g1[col >> 5]);
        #pragma unroll
        for (int i = 0; i < 8; ++i) {
            int rowl = m * 16 + i + hi * 8;
            X1[rowl * XSTRIDE + col] = (__bf16)(fast_tanhf(c[i] + bias) * gate);
        }
    }
    __syncthreads();

    // ---- stage 2: X2 = gate2 * tanh( X1[64x1024] @ W2'[1024x1024] + b2 ) ---
    for (int pass = 0; pass < 4; ++pass) {
        int n0 = pass * 16 + wave * 2;              // each wave: 4 M x 2 N tiles
        v8f acc[4][2];
        #pragma unroll
        for (int m = 0; m < 4; ++m) { acc[m][0] = (v8f){0,0,0,0,0,0,0,0}; acc[m][1] = acc[m][0]; }
        for (int k = 0; k < 32; ++k) {
            AFrag bf0, bf1;
            const uint4* p0 = (const uint4*)(pW2 + (((long)n0       * 32 + k) * 512 + lane * 16));
            const uint4* p1 = (const uint4*)(pW2 + (((long)(n0 + 1) * 32 + k) * 512 + lane * 16));
            bf0.u[0] = p0[0]; bf0.u[1] = p0[1];
            bf1.u[0] = p1[0]; bf1.u[1] = p1[1];
            #pragma unroll
            for (int m = 0; m < 4; ++m) {
                AFrag a;
                const unsigned char* abase = (const unsigned char*)X1
                    + (m * 16 + r) * (XSTRIDE * 2) + k * 64 + hi * 16;
                a.u[0] = *(const uint4*)(abase);
                a.u[1] = *(const uint4*)(abase + 32);
                acc[m][0] = __builtin_amdgcn_wmma_f32_16x16x32_bf16(false, a.v, false, bf0.v, (short)0, acc[m][0], false, false);
                acc[m][1] = __builtin_amdgcn_wmma_f32_16x16x32_bf16(false, a.v, false, bf1.v, (short)0, acc[m][1], false, false);
            }
        }
        #pragma unroll
        for (int j = 0; j < 2; ++j) {
            int col = (n0 + j) * 16 + r;
            float bias = b2[col];
            float gate = fast_sigmoid(g2[col >> 5]);
            #pragma unroll
            for (int m = 0; m < 4; ++m)
                #pragma unroll
                for (int i = 0; i < 8; ++i) {
                    int rowl = m * 16 + i + hi * 8;
                    X2[rowl * XSTRIDE + col] = (__bf16)(fast_tanhf(acc[m][j][i] + bias) * gate);
                }
        }
    }
    __syncthreads();

    // ---- stage 3: H1f = X2[64x1024] @ Wq1[1024x256] + bq1 ------------------
    {
        int n0 = wave * 2;                          // 16 N tiles / 8 waves
        v8f acc[4][2];
        #pragma unroll
        for (int m = 0; m < 4; ++m) { acc[m][0] = (v8f){0,0,0,0,0,0,0,0}; acc[m][1] = acc[m][0]; }
        for (int k = 0; k < 32; ++k) {
            AFrag bf0, bf1;
            const uint4* p0 = (const uint4*)(pWq1 + (((long)n0       * 32 + k) * 512 + lane * 16));
            const uint4* p1 = (const uint4*)(pWq1 + (((long)(n0 + 1) * 32 + k) * 512 + lane * 16));
            bf0.u[0] = p0[0]; bf0.u[1] = p0[1];
            bf1.u[0] = p1[0]; bf1.u[1] = p1[1];
            #pragma unroll
            for (int m = 0; m < 4; ++m) {
                AFrag a;
                const unsigned char* abase = (const unsigned char*)X2
                    + (m * 16 + r) * (XSTRIDE * 2) + k * 64 + hi * 16;
                a.u[0] = *(const uint4*)(abase);
                a.u[1] = *(const uint4*)(abase + 32);
                acc[m][0] = __builtin_amdgcn_wmma_f32_16x16x32_bf16(false, a.v, false, bf0.v, (short)0, acc[m][0], false, false);
                acc[m][1] = __builtin_amdgcn_wmma_f32_16x16x32_bf16(false, a.v, false, bf1.v, (short)0, acc[m][1], false, false);
            }
        }
        #pragma unroll
        for (int j = 0; j < 2; ++j) {
            int col = (n0 + j) * 16 + r;
            float bias = bq1[col];
            #pragma unroll
            for (int m = 0; m < 4; ++m)
                #pragma unroll
                for (int i = 0; i < 8; ++i) {
                    int rowl = m * 16 + i + hi * 8;
                    H1f[rowl * 256 + col] = acc[m][j][i] + bias;
                }
        }
    }
    __syncthreads();

    // ---- stage 4: LayerNorm1 + ReLU -> H1 (bf16), 4 threads per row --------
    {
        const int rrow = tid >> 2, ch = tid & 3;    // 64 rows x 4 chunks of 64
        const float* rp = H1f + rrow * 256 + ch * 64;
        float sum = 0.f, sq = 0.f;
        #pragma unroll 8
        for (int j = 0; j < 64; ++j) { float v = rp[j]; sum += v; sq += v * v; }
        red[(rrow * 4 + ch) * 2 + 0] = sum;
        red[(rrow * 4 + ch) * 2 + 1] = sq;
        __syncthreads();
        if (ch == 0) {
            float s = 0.f, q = 0.f;
            #pragma unroll
            for (int c = 0; c < 4; ++c) { s += red[(rrow*4+c)*2]; q += red[(rrow*4+c)*2+1]; }
            float mu = s * (1.0f / 256.0f);
            float var = q * (1.0f / 256.0f) - mu * mu;
            stats[rrow * 2 + 0] = mu;
            stats[rrow * 2 + 1] = rsqrtf(var + 1e-5f);
        }
        __syncthreads();
        float mu = stats[rrow * 2 + 0], rs = stats[rrow * 2 + 1];
        #pragma unroll 8
        for (int j = 0; j < 64; ++j) {
            int j2 = ch * 64 + j;
            float v = (rp[j] - mu) * rs * ln1g[j2] + ln1b[j2];
            H1[rrow * H1STRIDE + j2] = (__bf16)fmaxf(v, 0.0f);
        }
    }
    __syncthreads();

    // ---- stage 5: H2f = H1[64x256] @ Wq2[256x128] + bq2 --------------------
    {
        int n = wave;                               // 8 N tiles / 8 waves
        v8f acc[4];
        #pragma unroll
        for (int m = 0; m < 4; ++m) acc[m] = (v8f){0,0,0,0,0,0,0,0};
        for (int k = 0; k < 8; ++k) {
            AFrag b;
            const uint4* p = (const uint4*)(pWq2 + (((long)n * 8 + k) * 512 + lane * 16));
            b.u[0] = p[0]; b.u[1] = p[1];
            #pragma unroll
            for (int m = 0; m < 4; ++m) {
                AFrag a;
                const unsigned char* abase = (const unsigned char*)H1
                    + (m * 16 + r) * (H1STRIDE * 2) + k * 64 + hi * 16;
                a.u[0] = *(const uint4*)(abase);
                a.u[1] = *(const uint4*)(abase + 32);
                acc[m] = __builtin_amdgcn_wmma_f32_16x16x32_bf16(false, a.v, false, b.v, (short)0, acc[m], false, false);
            }
        }
        int col = n * 16 + r;
        float bias = bq2[col];
        #pragma unroll
        for (int m = 0; m < 4; ++m)
            #pragma unroll
            for (int i = 0; i < 8; ++i) {
                int rowl = m * 16 + i + hi * 8;
                H2f[rowl * 128 + col] = acc[m][i] + bias;
            }
    }
    __syncthreads();

    // ---- stage 6: LayerNorm2 + ReLU + dot(Wq3) + bq3 -> out, 4 thr/row -----
    {
        const int rrow = tid >> 2, ch = tid & 3;    // 64 rows x 4 chunks of 32
        const float* rp = H2f + rrow * 128 + ch * 32;
        float sum = 0.f, sq = 0.f;
        #pragma unroll 8
        for (int j = 0; j < 32; ++j) { float v = rp[j]; sum += v; sq += v * v; }
        red[(rrow * 4 + ch) * 2 + 0] = sum;
        red[(rrow * 4 + ch) * 2 + 1] = sq;
        __syncthreads();
        if (ch == 0) {
            float s = 0.f, q = 0.f;
            #pragma unroll
            for (int c = 0; c < 4; ++c) { s += red[(rrow*4+c)*2]; q += red[(rrow*4+c)*2+1]; }
            float mu = s * (1.0f / 128.0f);
            float var = q * (1.0f / 128.0f) - mu * mu;
            stats[rrow * 2 + 0] = mu;
            stats[rrow * 2 + 1] = rsqrtf(var + 1e-5f);
        }
        __syncthreads();
        float mu = stats[rrow * 2 + 0], rs = stats[rrow * 2 + 1];
        float part = 0.f;
        #pragma unroll 8
        for (int j = 0; j < 32; ++j) {
            int j2 = ch * 32 + j;
            float v = (rp[j] - mu) * rs * ln2g[j2] + ln2b[j2];
            part += fmaxf(v, 0.0f) * Wq3[j2];
        }
        red[(rrow * 4 + ch) * 2 + 0] = part;
        __syncthreads();
        if (ch == 0) {
            float accv = 0.f;
            #pragma unroll
            for (int c = 0; c < 4; ++c) accv += red[(rrow*4+c)*2];
            out[row0 + rrow] = accv + bq3[0];
        }
    }
}

extern "C" void kernel_launch(void* const* d_in, const int* in_sizes, int n_in,
                              void* d_out, int out_size, void* d_ws, size_t ws_size,
                              hipStream_t stream) {
    const float* state = (const float*)d_in[0];
    const float* action = (const float*)d_in[1];
    const float* W1  = (const float*)d_in[2];
    const float* b1  = (const float*)d_in[3];
    const float* g1  = (const float*)d_in[4];
    const float* W2  = (const float*)d_in[5];
    const float* b2  = (const float*)d_in[6];
    const float* g2  = (const float*)d_in[7];
    const float* Wq1 = (const float*)d_in[8];
    const float* bq1 = (const float*)d_in[9];
    const float* ln1g = (const float*)d_in[10];
    const float* ln1b = (const float*)d_in[11];
    const float* Wq2 = (const float*)d_in[12];
    const float* bq2 = (const float*)d_in[13];
    const float* ln2g = (const float*)d_in[14];
    const float* ln2b = (const float*)d_in[15];
    const float* Wq3 = (const float*)d_in[16];
    const float* bq3 = (const float*)d_in[17];

    __bf16* pW1  = (__bf16*)d_ws;            // 32768 bf16
    __bf16* pW2  = pW1  + 32768;             // 1048576 bf16
    __bf16* pWq1 = pW2  + 1048576;           // 262144 bf16
    __bf16* pWq2 = pWq1 + 262144;            // 32768 bf16

    // total packed elements = 1376256 -> 5376 blocks of 256
    pack_weights<<<5376, 256, 0, stream>>>(W1, W2, Wq1, Wq2, pW1, pW2, pWq1, pWq2);

    const int Bsz = in_sizes[0] / 24;        // 131072
    critic_fused<<<Bsz / ROWS, 256, 0, stream>>>(
        state, action, b1, g1, b2, g2, bq1, ln1g, ln1b, bq2, ln2g, ln2b,
        Wq3, bq3, pW1, pW2, pWq1, pWq2, (float*)d_out);
}
